// RetNet_42597485642448
// MI455X (gfx1250) — matrix operand order repaired and checked by reading, not compiled, not run on hardware
//
#include <hip/hip_runtime.h>
#include <hip/hip_bf16.h>
#include <math.h>

#define D_   1024
#define H_   16
#define DK_  64
#define DV_  128
#define L_   3
#define FFN_ 4096
#define S_   1024
#define B_   4
#define V_   2
#define M_   (B_*S_)

typedef _Float16 h8   __attribute__((ext_vector_type(8)));
typedef _Float16 h16  __attribute__((ext_vector_type(16)));
typedef float    f8   __attribute__((ext_vector_type(8)));

union Frag { h16 v; h8 h[2]; };

static __device__ __forceinline__ f8 wmma_f16(const Frag& a, const Frag& b, f8 c) {
    return __builtin_amdgcn_wmma_f32_16x16x32_f16(false, a.v, false, b.v, (short)0, c, false, false);
}

// gfx1250 async global->LDS copy (CDNA5 ISA): 16B per lane,
// tracked with ASYNCcnt.  ldsoff = LDS byte address, gp = 64-bit global address.
static __device__ __forceinline__ void async_b128(unsigned ldsoff, const void* gp) {
    asm volatile("global_load_async_to_lds_b128 %0, %1, off"
                 :: "v"(ldsoff), "v"(gp) : "memory");
}
static __device__ __forceinline__ void wait_async0() {
    asm volatile("s_wait_asynccnt 0x0" ::: "memory");
}

// ---------------------------------------------------------------- utilities
__global__ void k_cvt_wT(const float* __restrict__ W, _Float16* __restrict__ WT,
                         int K, int N) {
    size_t idx = (size_t)blockIdx.x * blockDim.x + threadIdx.x;
    size_t total = (size_t)K * N;
    if (idx >= total) return;
    int n = (int)(idx / K);
    int k = (int)(idx % K);
    WT[idx] = (_Float16)W[(size_t)k * N + n];
}

__global__ void k_embed(const int* __restrict__ tok, const float* __restrict__ emb,
                        float* __restrict__ hbuf) {
    size_t idx = (size_t)blockIdx.x * blockDim.x + threadIdx.x;
    if (idx >= (size_t)M_ * D_) return;
    int d  = (int)(idx % D_);
    int rs = (int)(idx / D_);
    int s  = rs % S_;
    int b  = rs / S_;
    int t  = (s == 0) ? 0 : tok[b * S_ + s - 1];
    hbuf[idx] = emb[(size_t)t * D_ + d] * 32.0f;   // sqrt(1024)
}

__global__ void k_ln_f16(const float* __restrict__ x, const float* __restrict__ sc,
                         const float* __restrict__ bi, _Float16* __restrict__ out) {
    __shared__ float red[256];
    int row = blockIdx.x, tid = threadIdx.x;
    const float* xr = x + (size_t)row * D_;
    float s = 0.f;
    for (int i = tid; i < D_; i += 256) s += xr[i];
    red[tid] = s; __syncthreads();
    for (int o = 128; o > 0; o >>= 1) { if (tid < o) red[tid] += red[tid + o]; __syncthreads(); }
    float mean = red[0] * (1.0f / D_); __syncthreads();
    float v = 0.f;
    for (int i = tid; i < D_; i += 256) { float d = xr[i] - mean; v += d * d; }
    red[tid] = v; __syncthreads();
    for (int o = 128; o > 0; o >>= 1) { if (tid < o) red[tid] += red[tid + o]; __syncthreads(); }
    float inv = rsqrtf(red[0] * (1.0f / D_) + 1e-5f);
    for (int i = tid; i < D_; i += 256)
        out[(size_t)row * D_ + i] = (_Float16)((xr[i] - mean) * inv * sc[i] + bi[i]);
}

// ---------------------------------------------------------------- GEMM (WMMA)
// C[M,N] = act(A[M,K] @ B[K,N] + bias) (+ resid);  B supplied transposed: BT[N,K].
// Block = 8 waves -> 128x64 output block.  Shared 64x32 B tile double-buffered in
// LDS, filled by global_load_async_to_lds_b128 (ASYNCcnt), consumed via ds_load.
#define BROW 40   // padded LDS row length in f16 (64B data + 16B pad -> 20 banks)
__global__ void __launch_bounds__(256) k_gemm(
        const _Float16* __restrict__ A, const _Float16* __restrict__ BT,
        const float* __restrict__ bias, const float* __restrict__ resid,
        float* __restrict__ Cf, _Float16* __restrict__ Ch,
        int Mm, int N, int K, int act) {
    __shared__ __align__(16) _Float16 btile[2][64 * BROW];
    int tid  = threadIdx.x;
    int lane = tid & 31;
    int wv   = tid >> 5;
    int nblocks = N >> 6;
    int mb = blockIdx.x / nblocks;
    int nb = blockIdx.x % nblocks;
    int r0 = mb * 128 + wv * 16;
    int n0 = nb << 6;
    int ra   = lane & 15;
    int half = lane >> 4;
    int ak = half ? 8 : 0;     // A-frag K group base (ISA 7.12.2 16-bit A layout)
    int bk = half ? 8 : 0;     // B-frag: same K groups as A-frag

    // per-thread share of the B-tile copy: 16B each, 256 threads = 64 rows x 64B
    int crow = tid >> 2;            // 0..63 : column n0+crow of B (row of BT)
    int cchk = (tid & 3) << 3;      // 0,8,16,24 f16 elements within the 32-K slab
    const _Float16* gsrc = BT + (size_t)(n0 + crow) * K + cchk;
    unsigned ldst0 = (unsigned)(size_t)&btile[0][crow * BROW + cchk];
    unsigned ldst1 = (unsigned)(size_t)&btile[1][crow * BROW + cchk];

    f8 acc[4];
    #pragma unroll
    for (int t = 0; t < 4; ++t) acc[t] = (f8)0.f;

    const _Float16* Arow = A + (size_t)(r0 + ra) * K;

    // prologue: stage K-slab 0
    async_b128(ldst0, gsrc);
    wait_async0();
    __syncthreads();

    int nk = K >> 5;
    for (int ki = 0; ki < nk; ++ki) {
        int cur = ki & 1;
        if (ki + 1 < nk)                       // stage next slab while computing
            async_b128(cur ? ldst0 : ldst1, gsrc + ((ki + 1) << 5));
        int k0 = ki << 5;
        if (k0 + 128 < K) __builtin_prefetch(Arow + k0 + 128, 0, 3);
        Frag a;
        a.h[0] = *(const h8*)(Arow + k0 + ak);
        a.h[1] = *(const h8*)(Arow + k0 + ak + 16);
        #pragma unroll
        for (int t = 0; t < 4; ++t) {
            const _Float16* bp = &btile[cur][(t * 16 + ra) * BROW + bk];
            Frag b;
            b.h[0] = *(const h8*)bp;
            b.h[1] = *(const h8*)(bp + 16);
            acc[t] = wmma_f16(a, b, acc[t]);
        }
        wait_async0();      // next slab finished (overlapped with WMMAs above)
        __syncthreads();    // publish to all waves / all waves done reading cur
    }

    #pragma unroll
    for (int t = 0; t < 4; ++t) {
        int n = n0 + t * 16 + ra;
        float bb = bias ? bias[n] : 0.f;
        #pragma unroll
        for (int vg = 0; vg < 8; ++vg) {
            int m = r0 + vg + (half << 3);       // C layout: lanes>=16 hold M+8
            float val = acc[t][vg] + bb;
            if (act) {
                float xg = val;
                val = 0.5f * xg * (1.f + tanhf(0.7978845608028654f * (xg + 0.044715f * xg * xg * xg)));
            }
            size_t idx = (size_t)m * N + n;
            if (resid) val += resid[idx];
            if (Cf) Cf[idx] = val;
            if (Ch) Ch[idx] = (_Float16)val;
        }
    }
}

// ---------------------------------------------------------------- RoPE + pack
// in: f32 [b,s,H*DK] ; out: f16 [b,h,s,dk]
__global__ void k_rope(const float* __restrict__ in, _Float16* __restrict__ out, float scale) {
    size_t idx = (size_t)blockIdx.x * blockDim.x + threadIdx.x;
    if (idx >= (size_t)M_ * H_ * (DK_ / 2)) return;
    int p  = (int)(idx % (DK_ / 2));
    int r  = (int)(idx / (DK_ / 2));
    int hh = r % H_; int r2 = r / H_;
    int s  = r2 % S_; int b = r2 / S_;
    float ang = powf(10000.f, -(float)p / 31.f);
    float sn, cs; sincosf((float)s * ang, &sn, &cs);
    size_t ib = ((size_t)(b * S_ + s) * H_ + hh) * DK_;
    float x0 = in[ib + 2 * p] * scale;
    float x1 = in[ib + 2 * p + 1] * scale;
    size_t ob = ((size_t)(b * H_ + hh) * S_ + s) * DK_;
    out[ob + 2 * p]     = (_Float16)(x0 * cs - x1 * sn);
    out[ob + 2 * p + 1] = (_Float16)(x1 * cs + x0 * sn);
}

// in: f32 [b,s,H*DV] ; out: f16 [b,h,dv,s]
__global__ void k_packvT(const float* __restrict__ in, _Float16* __restrict__ out) {
    size_t idx = (size_t)blockIdx.x * blockDim.x + threadIdx.x;
    if (idx >= (size_t)B_ * H_ * DV_ * S_) return;
    int s  = (int)(idx % S_); size_t r = idx / S_;
    int dv = (int)(r % DV_);  size_t r2 = r / DV_;
    int hh = (int)(r2 % H_);  int b = (int)(r2 / H_);
    out[idx] = (_Float16)in[((size_t)(b * S_ + s) * H_ + hh) * DV_ + dv];
}

// ---------------------------------------------------------------- attention
// Q,K f16 [b,h,s,dk]; Vt f16 [b,h,dv,s]; O f32 [b,s,H*DV]. One wave per (b,h,16-row tile).
__global__ void k_attn(const _Float16* __restrict__ Q, const _Float16* __restrict__ Kt,
                       const _Float16* __restrict__ Vt, float* __restrict__ O) {
    __shared__ __align__(16) _Float16 alds[16 * 32];
    int lane = threadIdx.x;
    int wid  = blockIdx.x;
    int ntile = wid % (S_ / 16); wid /= (S_ / 16);
    int hh = wid % H_; int b = wid / H_;
    int n0 = ntile * 16;
    const _Float16* q  = Q  + (size_t)(b * H_ + hh) * S_ * DK_;
    const _Float16* kk = Kt + (size_t)(b * H_ + hh) * S_ * DK_;
    const _Float16* vv = Vt + (size_t)(b * H_ + hh) * DV_ * S_;

    float gamma = 1.f - exp2f(-5.f - (float)hh);
    float lg    = log2f(gamma);
    float invg  = 1.f / (1.f - gamma);

    int ra = lane & 15, half = lane >> 4;
    int ak = half ? 8 : 0;
    int bk = half ? 8 : 0;

    Frag qa0, qa1;
    {
        const _Float16* qrow = q + (size_t)(n0 + ra) * DK_;
        qa0.h[0] = *(const h8*)(qrow + ak);
        qa0.h[1] = *(const h8*)(qrow + ak + 16);
        qa1.h[0] = *(const h8*)(qrow + 32 + ak);
        qa1.h[1] = *(const h8*)(qrow + 32 + ak + 16);
    }

    f8 oacc[8];
    #pragma unroll
    for (int c = 0; c < 8; ++c) oacc[c] = (f8)0.f;
    float dpart[8];
    #pragma unroll
    for (int vg = 0; vg < 8; ++vg) dpart[vg] = 0.f;

    int m0max = ((n0 + 15) >> 5) << 5;
    for (int m0 = 0; m0 <= m0max; m0 += 32) {
        f8 qk[2];
        #pragma unroll
        for (int st = 0; st < 2; ++st) {
            const _Float16* krow = kk + (size_t)(m0 + st * 16 + ra) * DK_;
            Frag b0, b1;
            b0.h[0] = *(const h8*)(krow + bk);
            b0.h[1] = *(const h8*)(krow + bk + 16);
            b1.h[0] = *(const h8*)(krow + 32 + bk);
            b1.h[1] = *(const h8*)(krow + 32 + bk + 16);
            f8 a = (f8)0.f;
            a = wmma_f16(qa0, b0, a);
            a = wmma_f16(qa1, b1, a);
            qk[st] = a;
        }
        #pragma unroll
        for (int st = 0; st < 2; ++st) {
            #pragma unroll
            for (int vg = 0; vg < 8; ++vg) {
                int n = n0 + vg + (half << 3);
                int m = m0 + st * 16 + ra;
                float dec = 0.f;
                if (m <= n) {
                    float dsum = (1.f - exp2f((float)(n + 1) * lg)) * invg;
                    dec = exp2f((float)(n - m) * lg) * rsqrtf(dsum);
                }
                float aval = qk[st][vg] * dec;
                dpart[vg] += fabsf(aval);
                alds[(vg + (half << 3)) * 32 + st * 16 + ra] = (_Float16)aval;
            }
        }
        __syncthreads();
        Frag af;
        af.h[0] = *(const h8*)(&alds[ra * 32 + ak]);
        af.h[1] = *(const h8*)(&alds[ra * 32 + ak + 16]);
        #pragma unroll
        for (int c = 0; c < 8; ++c) {
            const _Float16* vcol = vv + (size_t)(c * 16 + ra) * S_ + m0 + bk;
            Frag vb;
            vb.h[0] = *(const h8*)vcol;
            vb.h[1] = *(const h8*)(vcol + 16);
            oacc[c] = wmma_f16(af, vb, oacc[c]);
        }
        __syncthreads();
    }
    float invden[8];
    #pragma unroll
    for (int vg = 0; vg < 8; ++vg) {
        float d = dpart[vg];
        d += __shfl_xor(d, 1, 32); d += __shfl_xor(d, 2, 32);
        d += __shfl_xor(d, 4, 32); d += __shfl_xor(d, 8, 32);
        invden[vg] = 1.f / fmaxf(d, 1.f);
    }
    float scl[8];
    #pragma unroll
    for (int vg = 0; vg < 8; ++vg) {
        float ssq = 0.f;
        #pragma unroll
        for (int c = 0; c < 8; ++c) {
            float v = oacc[c][vg] * invden[vg];
            ssq += v * v;
        }
        ssq += __shfl_xor(ssq, 1, 32); ssq += __shfl_xor(ssq, 2, 32);
        ssq += __shfl_xor(ssq, 4, 32); ssq += __shfl_xor(ssq, 8, 32);
        scl[vg] = rsqrtf(ssq * (1.0f / DV_) + 1e-6f);
    }
    #pragma unroll
    for (int c = 0; c < 8; ++c) {
        #pragma unroll
        for (int vg = 0; vg < 8; ++vg) {
            int n  = n0 + vg + (half << 3);
            int dv = c * 16 + ra;
            O[((size_t)(b * S_ + n)) * (H_ * DV_) + hh * DV_ + dv] =
                oacc[c][vg] * invden[vg] * scl[vg];
        }
    }
}

__global__ void k_gate(const float* __restrict__ g, const float* __restrict__ o,
                       _Float16* __restrict__ og) {
    size_t idx = (size_t)blockIdx.x * blockDim.x + threadIdx.x;
    if (idx >= (size_t)M_ * 2 * D_) return;
    float gv = g[idx];
    float sv = gv / (1.f + expf(-gv));
    og[idx] = (_Float16)(sv * o[idx]);
}

// final LN + logits(D x 2) + log_softmax
__global__ void k_final(const float* __restrict__ hbuf, const float* __restrict__ sc,
                        const float* __restrict__ bi, const float* __restrict__ Wout,
                        float* __restrict__ out) {
    __shared__ float red[256];
    int row = blockIdx.x, tid = threadIdx.x;
    const float* xr = hbuf + (size_t)row * D_;
    float s = 0.f;
    for (int i = tid; i < D_; i += 256) s += xr[i];
    red[tid] = s; __syncthreads();
    for (int o = 128; o > 0; o >>= 1) { if (tid < o) red[tid] += red[tid + o]; __syncthreads(); }
    float mean = red[0] * (1.0f / D_); __syncthreads();
    float v = 0.f;
    for (int i = tid; i < D_; i += 256) { float d = xr[i] - mean; v += d * d; }
    red[tid] = v; __syncthreads();
    for (int o = 128; o > 0; o >>= 1) { if (tid < o) red[tid] += red[tid + o]; __syncthreads(); }
    float inv = rsqrtf(red[0] * (1.0f / D_) + 1e-5f);
    __syncthreads();
    float d0 = 0.f, d1 = 0.f;
    for (int i = tid; i < D_; i += 256) {
        float z = (xr[i] - mean) * inv * sc[i] + bi[i];
        d0 += z * Wout[(size_t)i * V_];
        d1 += z * Wout[(size_t)i * V_ + 1];
    }
    red[tid] = d0; __syncthreads();
    for (int o = 128; o > 0; o >>= 1) { if (tid < o) red[tid] += red[tid + o]; __syncthreads(); }
    float l0 = red[0]; __syncthreads();
    red[tid] = d1; __syncthreads();
    for (int o = 128; o > 0; o >>= 1) { if (tid < o) red[tid] += red[tid + o]; __syncthreads(); }
    float l1 = red[0];
    if (tid == 0) {
        float mx  = fmaxf(l0, l1);
        float lse = mx + logf(expf(l0 - mx) + expf(l1 - mx));
        out[(size_t)row * V_]     = l0 - lse;
        out[(size_t)row * V_ + 1] = l1 - lse;
    }
}

// ---------------------------------------------------------------- host
static inline int g1(size_t n) { return (int)((n + 255) / 256); }

extern "C" void kernel_launch(void* const* d_in, const int* in_sizes, int n_in,
                              void* d_out, int out_size, void* d_ws, size_t ws_size,
                              hipStream_t stream) {
    const int*   tokens = (const int*)  d_in[0];
    const float* emb    = (const float*)d_in[1];
    const float* Wq     = (const float*)d_in[2];
    const float* Wk     = (const float*)d_in[3];
    const float* Wv     = (const float*)d_in[4];
    const float* Wg     = (const float*)d_in[5];
    const float* Wo     = (const float*)d_in[6];
    const float* ln1_s  = (const float*)d_in[7];
    const float* ln1_b  = (const float*)d_in[8];
    const float* ln2_s  = (const float*)d_in[9];
    const float* ln2_b  = (const float*)d_in[10];
    const float* W1     = (const float*)d_in[11];
    const float* b1     = (const float*)d_in[12];
    const float* W2     = (const float*)d_in[13];
    const float* b2     = (const float*)d_in[14];
    const float* lnf_s  = (const float*)d_in[15];
    const float* lnf_b  = (const float*)d_in[16];
    const float* Wout   = (const float*)d_in[17];
    float* out = (float*)d_out;

    char* p = (char*)d_ws;
    auto alloc = [&](size_t bytes) -> char* {
        char* r = p; p += (bytes + 255) & ~((size_t)255); return r;
    };
    float*    hB   = (float*)   alloc((size_t)M_ * D_ * 4);
    _Float16* z16  = (_Float16*)alloc((size_t)M_ * D_ * 2);
    float*    qf   = (float*)   alloc((size_t)M_ * D_ * 4);
    float*    kf   = (float*)   alloc((size_t)M_ * D_ * 4);
    float*    vf   = (float*)   alloc((size_t)M_ * 2 * D_ * 4);
    float*    gf   = (float*)   alloc((size_t)M_ * 2 * D_ * 4);
    float*    of   = (float*)   alloc((size_t)M_ * 2 * D_ * 4);
    _Float16* q16  = (_Float16*)alloc((size_t)M_ * D_ * 2);
    _Float16* k16  = (_Float16*)alloc((size_t)M_ * D_ * 2);
    _Float16* vT16 = (_Float16*)alloc((size_t)M_ * 2 * D_ * 2);
    _Float16* og16 = (_Float16*)alloc((size_t)M_ * 2 * D_ * 2);
    _Float16* ff16 = (_Float16*)alloc((size_t)M_ * FFN_ * 2);
    _Float16* wT   = (_Float16*)alloc((size_t)FFN_ * D_ * 2);  // reused per GEMM (max 8MB)

    auto gemm = [&](const _Float16* A, const _Float16* BT, const float* bias,
                    const float* resid, float* Cf, _Float16* Ch,
                    int Mm, int N, int K, int act) {
        int blocks = (Mm / 128) * (N / 64);   // M is always 4096 here -> exact
        k_gemm<<<blocks, 256, 0, stream>>>(A, BT, bias, resid, Cf, Ch, Mm, N, K, act);
    };
    auto cvt = [&](const float* W, int K, int N) {
        k_cvt_wT<<<g1((size_t)K * N), 256, 0, stream>>>(W, wT, K, N);
    };

    k_embed<<<g1((size_t)M_ * D_), 256, 0, stream>>>(tokens, emb, hB);

    for (int l = 0; l < L_; ++l) {
        // MSR block
        k_ln_f16<<<M_, 256, 0, stream>>>(hB, ln1_s + l * D_, ln1_b + l * D_, z16);
        cvt(Wq + (size_t)l * D_ * D_, D_, D_);
        gemm(z16, wT, nullptr, nullptr, qf, nullptr, M_, D_, D_, 0);
        cvt(Wk + (size_t)l * D_ * D_, D_, D_);
        gemm(z16, wT, nullptr, nullptr, kf, nullptr, M_, D_, D_, 0);
        cvt(Wv + (size_t)l * D_ * 2 * D_, D_, 2 * D_);
        gemm(z16, wT, nullptr, nullptr, vf, nullptr, M_, 2 * D_, D_, 0);
        cvt(Wg + (size_t)l * D_ * 2 * D_, D_, 2 * D_);
        gemm(z16, wT, nullptr, nullptr, gf, nullptr, M_, 2 * D_, D_, 0);

        k_rope<<<g1((size_t)M_ * H_ * (DK_ / 2)), 256, 0, stream>>>(qf, q16, 1.0f);
        k_rope<<<g1((size_t)M_ * H_ * (DK_ / 2)), 256, 0, stream>>>(kf, k16, 0.125f); // DK^-0.5
        k_packvT<<<g1((size_t)B_ * H_ * DV_ * S_), 256, 0, stream>>>(vf, vT16);

        k_attn<<<B_ * H_ * (S_ / 16), 32, 0, stream>>>(q16, k16, vT16, of);

        k_gate<<<g1((size_t)M_ * 2 * D_), 256, 0, stream>>>(gf, of, og16);
        cvt(Wo + (size_t)l * 2 * D_ * D_, 2 * D_, D_);
        gemm(og16, wT, nullptr, hB, hB, nullptr, M_, D_, 2 * D_, 0);   // h += msr(...)

        // FFN block
        k_ln_f16<<<M_, 256, 0, stream>>>(hB, ln2_s + l * D_, ln2_b + l * D_, z16);
        cvt(W1 + (size_t)l * D_ * FFN_, D_, FFN_);
        gemm(z16, wT, b1 + (size_t)l * FFN_, nullptr, nullptr, ff16, M_, FFN_, D_, 1); // gelu
        cvt(W2 + (size_t)l * FFN_ * D_, FFN_, D_);
        gemm(ff16, wT, b2 + (size_t)l * D_, hB, hB, nullptr, M_, D_, FFN_, 0);         // h += ffn
    }

    k_final<<<M_, 256, 0, stream>>>(hB, lnf_s, lnf_b, Wout, out);
}
